// Self_Attn_30966714204777
// MI455X (gfx1250) — matrix operand order repaired and checked by reading, not compiled, not run on hardware
//
#include <hip/hip_runtime.h>
#include <cstddef>
#include <cstdint>

#define BB   4
#define CC   256
#define NPIX 4096
#define DQK  32
#define KB   64        // keys per pipeline stage

typedef _Float16 half_t;
typedef __attribute__((ext_vector_type(8)))  _Float16     v8h;
typedef __attribute__((ext_vector_type(16))) _Float16     v16h;
typedef __attribute__((ext_vector_type(8)))  float        v8f;
typedef __attribute__((ext_vector_type(4)))  unsigned int v4u;
typedef __attribute__((ext_vector_type(8)))  int          v8i;
typedef __attribute__((ext_vector_type(4)))  int          v4i;

// ---------------------------------------------------------------------------
// VALU cross-lane xor-reductions within each 16-lane row via v_permlane16_b32
// ---------------------------------------------------------------------------

__device__ __forceinline__ float permx(float v, unsigned s0, unsigned s1) {
  int i = __builtin_bit_cast(int, v);
  i = __builtin_amdgcn_permlane16(i, i, (int)s0, (int)s1, false, false);
  return __builtin_bit_cast(float, i);
}
__device__ __forceinline__ float rowmax16(float v) {
  v = fmaxf(v, permx(v, 0x67452301u, 0xEFCDAB89u));   // xor 1
  v = fmaxf(v, permx(v, 0x54761032u, 0xDCFE98BAu));   // xor 2
  v = fmaxf(v, permx(v, 0x32107654u, 0xBA98FEDCu));   // xor 4
  v = fmaxf(v, permx(v, 0xFEDCBA98u, 0x76543210u));   // xor 8
  return v;
}
__device__ __forceinline__ float rowsum16(float v) {
  v += permx(v, 0x67452301u, 0xEFCDAB89u);
  v += permx(v, 0x54761032u, 0xDCFE98BAu);
  v += permx(v, 0x32107654u, 0xBA98FEDCu);
  v += permx(v, 0xFEDCBA98u, 0x76543210u);
  return v;
}

// ---------------------------------------------------------------------------
// Weight prep: pack Wq|Wk|Wv -> f16 Whf[320][256] (row-major), biases -> Bf[320]
// ---------------------------------------------------------------------------

__global__ __launch_bounds__(256)
void prep_w_kernel(const float* __restrict__ Wq, const float* __restrict__ bq,
                   const float* __restrict__ Wk, const float* __restrict__ bk,
                   const float* __restrict__ Wv, const float* __restrict__ bv,
                   half_t* __restrict__ Whf, float* __restrict__ Bf)
{
  const int r = blockIdx.x;       // 0..319 output row
  const int c = threadIdx.x;      // 0..255 input channel
  const float* src;
  float bias;
  if (r < 32)      { src = Wq + (size_t)r * CC;        bias = bq[r];      }
  else if (r < 64) { src = Wk + (size_t)(r - 32) * CC; bias = bk[r - 32]; }
  else             { src = Wv + (size_t)(r - 64) * CC; bias = bv[r - 64]; }
  Whf[(size_t)r * CC + c] = (half_t)src[c];
  if (c == 0) Bf[r] = bias;
}

// ---------------------------------------------------------------------------
// WMMA projection GEMM: per batch (N x 256) * (256 x 320) -> Qh, Kh, Vh
// Block = 4 waves, 64 pixels; x tile staged to LDS (transposed, f16).
// ---------------------------------------------------------------------------

__global__ __launch_bounds__(128)
void proj_wmma_kernel(const float* __restrict__ x,
                      const half_t* __restrict__ Whf,
                      const float* __restrict__ Bf,
                      half_t* __restrict__ Qh, half_t* __restrict__ Kh,
                      half_t* __restrict__ Vh)
{
  __shared__ __align__(16) half_t Xl[64][CC];          // 32 KB, pixel-major
  const int n0   = blockIdx.x * 64;
  const int b    = blockIdx.y;
  const int tid  = threadIdx.x;
  const int wave = tid >> 5;
  const int lane = tid & 31;
  const int m    = lane & 15;
  const int hi   = lane >> 4;

  // ---- stage + transpose + f16-convert x tile (coalesced along n)
  const float* xb = x + (size_t)b * CC * NPIX;
  for (int it = 0; it < 128; ++it) {
    const int idx = it * 128 + tid;                    // over 256*64
    const int c = idx >> 6, n = idx & 63;
    Xl[n][c] = (half_t)xb[(size_t)c * NPIX + n0 + n];
  }
  __syncthreads();

  // ---- A operand chunks (8 x v16h) for this wave's 16 pixels
  v16h A[8];
#pragma unroll
  for (int ch = 0; ch < 8; ++ch) {
    const half_t* ap = &Xl[16 * wave + m][32 * ch];
    v8h a0 = *(const v8h*)(ap + 16 * hi);
    v8h a1 = *(const v8h*)(ap + 16 * hi + 8);
#pragma unroll
    for (int e = 0; e < 8; ++e) { A[ch][e] = a0[e]; A[ch][e + 8] = a1[e]; }
  }

  const int prow = n0 + 16 * wave;                     // pixel tile base
  for (int t = 0; t < 20; ++t) {                       // 2 q + 2 k + 16 v tiles
    const int drow = 16 * t + m;                       // lane's output column
    v8f acc;
#pragma unroll
    for (int r = 0; r < 8; ++r) acc[r] = 0.0f;
#pragma unroll
    for (int ch = 0; ch < 8; ++ch) {
      v16h Bv = *(const v16h*)(&Whf[(size_t)drow * CC + 32 * ch + 16 * hi]);
      acc = __builtin_amdgcn_wmma_f32_16x16x32_f16(false, A[ch], false, Bv,
                                                   (short)0, acc, false, false);
    }
    const float bias = Bf[drow];
    if (t < 2) {                                       // Q: (B,N,32) pixel-major
      const int d = 16 * t + m;
#pragma unroll
      for (int r = 0; r < 8; ++r)
        Qh[((size_t)b * NPIX + prow + r + 8 * hi) * DQK + d] = (half_t)(acc[r] + bias);
    } else if (t < 4) {                                // K: (B,N,32) pixel-major
      const int d = 16 * (t - 2) + m;
#pragma unroll
      for (int r = 0; r < 8; ++r)
        Kh[((size_t)b * NPIX + prow + r + 8 * hi) * DQK + d] = (half_t)(acc[r] + bias);
    } else {                                           // V: (B,C,N) channel-major
      const int c = 16 * (t - 4) + m;
      v8h vo;
#pragma unroll
      for (int r = 0; r < 8; ++r) vo[r] = (half_t)(acc[r] + bias);
      *(v8h*)(&Vh[((size_t)b * CC + c) * NPIX + prow + 8 * hi]) = vo;
    }
  }
}

// ---------------------------------------------------------------------------
// TDM helper: 2-D tile load Global -> LDS.  D# per CDNA5 ISA.
// ---------------------------------------------------------------------------

__device__ __forceinline__ void tdm_load_2d(unsigned lds_off, const void* gptr,
                                            unsigned tensor_d0, unsigned tensor_d1,
                                            unsigned stride0,
                                            unsigned tile_d0, unsigned tile_d1)
{
  const unsigned long long ga = (unsigned long long)(uintptr_t)gptr;
  v4u g0;
  g0[0] = 1u;                                          // count = 1, user mode
  g0[1] = lds_off;                                     // lds_addr (bytes)
  g0[2] = (unsigned)ga;                                // global_addr[31:0]
  g0[3] = (unsigned)((ga >> 32) & 0x01FFFFFFu) | 0x80000000u;  // addr[56:32] | type=2
  v8i g1;
  g1[0] = 0x10000;                                     // data_size = 1 -> 2-byte elems
  g1[1] = (int)((tensor_d0 & 0xFFFFu) << 16);          // tensor_dim0[15:0]
  g1[2] = (int)((tensor_d0 >> 16) | ((tensor_d1 & 0xFFFFu) << 16));
  g1[3] = (int)((tensor_d1 >> 16) | (tile_d0 << 16));  // tile_dim0
  g1[4] = (int)(tile_d1 & 0xFFFFu);                    // tile_dim1; tile_dim2=0
  g1[5] = (int)stride0;                                // tensor_dim0_stride[31:0]
  g1[6] = 0;
  g1[7] = 0;
  v4i z4; z4[0] = 0; z4[1] = 0; z4[2] = 0; z4[3] = 0;
  v8i z8;
#pragma unroll
  for (int e = 0; e < 8; ++e) z8[e] = 0;
  __builtin_amdgcn_tensor_load_to_lds(g0, g1, z4, z4, z8, 0);
}

// ---------------------------------------------------------------------------
// Flash attention: TDM double-buffered K/V tiles (64 keys/stage),
// S = Q K^T (4 WMMA), online softmax, O += P V (32 WMMA)
// ---------------------------------------------------------------------------

__global__ __launch_bounds__(128)
void flash_attn_kernel(const half_t* __restrict__ Qh,
                       const half_t* __restrict__ Kh,
                       const half_t* __restrict__ Vh,
                       const float*  __restrict__ x,
                       const float*  __restrict__ gamma,
                       float* __restrict__ out)
{
  __shared__ __align__(16) half_t Kbuf[2][KB][DQK];    //  2 x 4 KB
  __shared__ __align__(16) half_t Vbuf[2][CC][KB];     //  2 x 32 KB
  __shared__ __align__(16) half_t Plds[4][16][KB];     //  8 KB per-wave P scratch

  const int wave = threadIdx.x >> 5;
  const int lane = threadIdx.x & 31;
  const int m    = lane & 15;                          // A-row / B-col / C-col index
  const int hi   = lane >> 4;                          // lane half-group
  const int b    = blockIdx.y;
  const int i0   = (blockIdx.x * 4 + wave) * 16;       // query tile base

  const half_t* Kb = Kh + (size_t)b * NPIX * DQK;
  const half_t* Vb = Vh + (size_t)b * CC * NPIX;

  // ---- Q tile in WMMA A layout
  const half_t* qrow = Qh + ((size_t)b * NPIX + (i0 + m)) * DQK;
  v8h q0 = *(const v8h*)(qrow + 16 * hi);
  v8h q1 = *(const v8h*)(qrow + 16 * hi + 8);
  v16h qa;
#pragma unroll
  for (int e = 0; e < 8; ++e) { qa[e] = q0[e]; qa[e + 8] = q1[e]; }

  // ---- accumulators: O over 16 channel tiles (C=256), per-row m/l state
  v8f O[16];
#pragma unroll
  for (int t = 0; t < 16; ++t)
#pragma unroll
    for (int r = 0; r < 8; ++r) O[t][r] = 0.0f;
  float mrow[8], lrow[8];
#pragma unroll
  for (int r = 0; r < 8; ++r) { mrow[r] = -__builtin_inff(); lrow[r] = 0.0f; }

  // ---- prologue: TDM-stage first K/V tile into buffer 0 (wave 0 only)
  if (wave == 0) {
    tdm_load_2d((unsigned)(uintptr_t)&Kbuf[0][0][0], Kb, DQK, NPIX, DQK, DQK, KB);
    tdm_load_2d((unsigned)(uintptr_t)&Vbuf[0][0][0], Vb, NPIX, CC, NPIX, KB, CC);
  }

  for (int j0 = 0; j0 < NPIX; j0 += KB) {
    const int cur = (j0 / KB) & 1;
    const int nxt = cur ^ 1;

    if (wave == 0) {
      if (j0 + KB < NPIX) {
        tdm_load_2d((unsigned)(uintptr_t)&Kbuf[nxt][0][0],
                    Kb + (size_t)(j0 + KB) * DQK, DQK, NPIX, DQK, DQK, KB);
        tdm_load_2d((unsigned)(uintptr_t)&Vbuf[nxt][0][0],
                    Vb + (j0 + KB), NPIX, CC, NPIX, KB, CC);
        __builtin_amdgcn_s_wait_tensorcnt(2);   // current buffer's 2 loads done
      } else {
        __builtin_amdgcn_s_wait_tensorcnt(0);
      }
    }
    __syncthreads();                            // LDS tiles visible to all waves

    // ---- S = Q K^T over 4 key sub-tiles
    v8f S[4];
#pragma unroll
    for (int i = 0; i < 4; ++i) {
      v16h kb = *(const v16h*)(&Kbuf[cur][16 * i + m][16 * hi]);
      v8f zero;
#pragma unroll
      for (int r = 0; r < 8; ++r) zero[r] = 0.0f;
      S[i] = __builtin_amdgcn_wmma_f32_16x16x32_f16(false, qa, false, kb,
                                                    (short)0, zero, false, false);
    }

    // ---- online softmax
    float scale[8];
    bool anych = false;
#pragma unroll
    for (int r = 0; r < 8; ++r) {
      float v = fmaxf(fmaxf(S[0][r], S[1][r]), fmaxf(S[2][r], S[3][r]));
      v = rowmax16(v);
      const float mn = fmaxf(mrow[r], v);
      scale[r] = __expf(mrow[r] - mn);
      anych |= (mn > mrow[r]);
      mrow[r] = mn;
    }

#pragma unroll
    for (int r = 0; r < 8; ++r) {
      float ps = 0.0f;
#pragma unroll
      for (int i = 0; i < 4; ++i) {
        const float p = __expf(S[i][r] - mrow[r]);
        ps += p;
        Plds[wave][r + 8 * hi][16 * i + m] = (half_t)p;
      }
      lrow[r] = lrow[r] * scale[r] + ps;
    }

    // ---- rescale running O only when a row max actually moved (wave-uniform)
    if (__builtin_amdgcn_ballot_w32(anych)) {
#pragma unroll
      for (int t = 0; t < 16; ++t)
#pragma unroll
        for (int r = 0; r < 8; ++r) O[t][r] *= scale[r];
    }

    asm volatile("" ::: "memory");

    // ---- reload P in WMMA A layout: two 16x32 A tiles
    const half_t* prow = &Plds[wave][m][0];
    v16h pa0, pa1;
    {
      v8h a0 = *(const v8h*)(prow + 16 * hi);
      v8h a1 = *(const v8h*)(prow + 16 * hi + 8);
      v8h a2 = *(const v8h*)(prow + 32 + 16 * hi);
      v8h a3 = *(const v8h*)(prow + 40 + 16 * hi);
#pragma unroll
      for (int e = 0; e < 8; ++e) {
        pa0[e] = a0[e]; pa0[e + 8] = a1[e];
        pa1[e] = a2[e]; pa1[e + 8] = a3[e];
      }
    }

    // ---- O += P * V over all 16 channel tiles
#pragma unroll
    for (int t = 0; t < 16; ++t) {
      v16h vb0 = *(const v16h*)(&Vbuf[cur][16 * t + m][16 * hi]);
      v16h vb1 = *(const v16h*)(&Vbuf[cur][16 * t + m][32 + 16 * hi]);
      O[t] = __builtin_amdgcn_wmma_f32_16x16x32_f16(false, pa0, false, vb0,
                                                    (short)0, O[t], false, false);
      O[t] = __builtin_amdgcn_wmma_f32_16x16x32_f16(false, pa1, false, vb1,
                                                    (short)0, O[t], false, false);
    }

    __syncthreads();                            // all reads done before overwrite
  }

  // ---- finalize: reduce l across half-group, out = gamma * O/l + x
  float linv[8];
#pragma unroll
  for (int r = 0; r < 8; ++r) linv[r] = 1.0f / rowsum16(lrow[r]);

  const float g = gamma[0];
#pragma unroll
  for (int t = 0; t < 16; ++t) {
    const int c = 16 * t + m;                   // C/D col -> channel
#pragma unroll
    for (int r = 0; r < 8; ++r) {
      const int i = i0 + r + 8 * hi;            // C/D row -> pixel
      const size_t idx = ((size_t)b * CC + c) * NPIX + i;
      out[idx] = g * (O[t][r] * linv[r]) + x[idx];
    }
  }
}

// ---------------------------------------------------------------------------

extern "C" void kernel_launch(void* const* d_in, const int* in_sizes, int n_in,
                              void* d_out, int out_size, void* d_ws, size_t ws_size,
                              hipStream_t stream) {
  const float* x     = (const float*)d_in[0];
  const float* Wq    = (const float*)d_in[1];
  const float* bq    = (const float*)d_in[2];
  const float* Wk    = (const float*)d_in[3];
  const float* bk    = (const float*)d_in[4];
  const float* Wv    = (const float*)d_in[5];
  const float* bv    = (const float*)d_in[6];
  const float* gamma = (const float*)d_in[7];
  float* out = (float*)d_out;

  // workspace: Qh 1MB | Kh 1MB | Vh 8MB | Whf 160KB | Bf 1.25KB  (f16/f32)
  half_t* Qh  = (half_t*)d_ws;
  half_t* Kh  = Qh + (size_t)BB * NPIX * DQK;
  half_t* Vh  = Kh + (size_t)BB * NPIX * DQK;
  half_t* Whf = Vh + (size_t)BB * CC * NPIX;
  float*  Bf  = (float*)(Whf + 320 * CC);

  prep_w_kernel<<<dim3(320), 256, 0, stream>>>(Wq, bq, Wk, bk, Wv, bv, Whf, Bf);
  proj_wmma_kernel<<<dim3(NPIX / 64, BB), 128, 0, stream>>>(x, Whf, Bf, Qh, Kh, Vh);
  flash_attn_kernel<<<dim3(NPIX / 64, BB), 128, 0, stream>>>(Qh, Kh, Vh, x, gamma, out);
}
